// NKF_12867722018949
// MI455X (gfx1250) — compile-verified, matrix operand order checked by reading, not run on hardware
//
#include <hip/hip_runtime.h>
#include <hip/hip_bf16.h>

#define NFFT  1024
#define HOP   256
#define LW    4
#define NBINS 513

typedef __attribute__((ext_vector_type(16))) __bf16 v16bf;
typedef __attribute__((ext_vector_type(8)))  __bf16 v8bf;
typedef __attribute__((ext_vector_type(8)))  float  v8f;

// ---------------------------------------------------------------- utilities
__device__ __forceinline__ float hannw(int i) {
  return 0.5f - 0.5f * cosf(6.28318530717958647692f * (float)i * (1.0f / 1024.0f));
}

// In-LDS 1024-point radix-2 DIT FFT, 256 threads.
__device__ void fft1024(float* re, float* im, bool inverse, int tid) {
  for (int i = tid; i < 1024; i += 256) {
    int r = __brev((unsigned)i) >> 22;          // reverse 10 bits
    if (r > i) {
      float tr = re[i]; re[i] = re[r]; re[r] = tr;
      float ti = im[i]; im[i] = im[r]; im[r] = ti;
    }
  }
  __syncthreads();
  for (int s = 1; s <= 10; ++s) {
    int mh = 1 << (s - 1);
    for (int b = tid; b < 512; b += 256) {
      int grp = b >> (s - 1);
      int pos = b & (mh - 1);
      int i0  = (grp << s) + pos;
      int i1  = i0 + mh;
      float ang = (inverse ? 6.28318530717958647692f : -6.28318530717958647692f)
                  * (float)pos / (float)(1 << s);
      float sn, cs;
      __sincosf(ang, &sn, &cs);
      float tr = re[i1] * cs - im[i1] * sn;
      float ti = re[i1] * sn + im[i1] * cs;
      float ar = re[i0], ai = im[i0];
      re[i1] = ar - tr; im[i1] = ai - ti;
      re[i0] = ar + tr; im[i0] = ai + ti;
    }
    __syncthreads();
  }
}

// ---------------------------------------------------------------- STFT
__global__ __launch_bounds__(256) void stft_kernel(
    const float* __restrict__ x, const float* __restrict__ y,
    float* __restrict__ Xr, float* __restrict__ Xi,
    float* __restrict__ Yr, float* __restrict__ Yi, int T, int LEN) {
  int t = blockIdx.x, b = blockIdx.y, sig = blockIdx.z, tid = threadIdx.x;
  const float* src = sig ? y : x;
  float* Pr = sig ? Yr : Xr;
  float* Pi = sig ? Yi : Xi;
  __shared__ float re[1024], im[1024];
  for (int i = tid; i < 1024; i += 256) {
    int s = t * HOP + i - 512;                 // reflect pad (center=True)
    if (s < 0) s = -s;
    if (s >= LEN) s = 2 * LEN - 2 - s;
    re[i] = src[(size_t)b * LEN + s] * hannw(i);
    im[i] = 0.f;
  }
  __syncthreads();
  fft1024(re, im, false, tid);
  for (int f = tid; f <= 512; f += 256) {
    size_t n = (size_t)(b * NBINS + f);
    Pr[n * T + t] = re[f];
    Pi[n * T + t] = im[f];
  }
}

// ---------------------------------------------------------------- skip flags
__global__ __launch_bounds__(256) void skip_kernel(
    const float* __restrict__ Xr, const float* __restrict__ Xi,
    int* __restrict__ skipf, int T, int N) {
  int t = blockIdx.x, tid = threadIdx.x;
  __shared__ float red[256];
  float s = 0.f;
  for (int idx = tid; idx < N * LW; idx += 256) {
    int n = idx >> 2, l = idx & 3;
    int tt = t - (LW - 1) + l;
    if (tt >= 0) {
      size_t o = (size_t)n * T + tt;
      float xr = Xr[o], xi = Xi[o];
      s += sqrtf(xr * xr + xi * xi);
    }
  }
  red[tid] = s;
  __syncthreads();
  for (int st = 128; st > 0; st >>= 1) {
    if (tid < st) red[tid] += red[tid + st];
    __syncthreads();
  }
  if (tid == 0) skipf[t] = (red[0] / (float)(N * LW) < 1e-5f) ? 1 : 0;
}

// ---------------------------------------------------------------- scan
struct ScanP {
  const float *a_in, *a_out;
  const float *fc1_iW, *fc1_ib, *fc1_rW, *fc1_rb;
  const float *fc2_iW, *fc2_ib, *fc2_rW, *fc2_rb;
  const float *fcin_iW, *fcin_ib, *fcin_rW, *fcin_rb;
  const float *gri_bhh, *gri_bih, *gri_whh, *gri_wih;
  const float *grr_bhh, *grr_bih, *grr_whh, *grr_wih;
  const float *Xr, *Xi, *Yr, *Yi;
  float *Er, *Ei;
  const int* skip;
  int T, N;
};

// K index for element j of a 16-element A/B lane payload (16x16x32 bf16 layout)
__device__ __forceinline__ int kidx(int j, int hf) {
  return (j < 8) ? (hf * 8 + j) : (8 + hf * 8 + j);   // j>=8 -> 16 + hf*8 + (j-8)
}

// A operand from bf16 canonical [16][32] LDS matrix: two contiguous 16B chunks.
__device__ __forceinline__ v16bf load_Abf(const __bf16* M, int lane) {
  int row = lane & 15, hf = lane >> 4;
  union { v16bf v; struct { v8bf lo, hi; } s; } u;
  u.s.lo = *(const v8bf*)(M + row * 32 + hf * 8);
  u.s.hi = *(const v8bf*)(M + row * 32 + 16 + hf * 8);
  return u.v;
}

__device__ __forceinline__ v8f wmma_bf(v16bf a, const __bf16* bsrc) {
  v16bf b = *(const v16bf*)bsrc;
  v8f c = {};
  return __builtin_amdgcn_wmma_f32_16x16x32_bf16(false, a, false, b, (short)0, c,
                                                 false, false);
}

// Scatter WMMA D (+bias) to f32 canonical matrix.
__device__ __forceinline__ void store_f32(v8f c, float* C, int cs, const float* bias,
                                          int c0, int Ocols, int lane) {
  int col = (lane & 15) + c0, hf = lane >> 4;
  if (col < Ocols) {
    float bv = bias[col];
#pragma unroll
    for (int r = 0; r < 8; ++r) C[(r + hf * 8) * cs + col] = c[r] + bv;
  }
}

// Scatter WMMA D (+bias, +PReLU) to bf16 canonical [16][32] matrix.
__device__ __forceinline__ void store_bf16(v8f c, __bf16* C, const float* bias,
                                           int c0, int Ocols, int lane, float alpha) {
  int col = (lane & 15) + c0, hf = lane >> 4;
  if (col < Ocols) {
    float bv = bias[col];
#pragma unroll
    for (int r = 0; r < 8; ++r) {
      float v = c[r] + bv;
      v = (v >= 0.f) ? v : alpha * v;
      C[(r + hf * 8) * 32 + col] = (__bf16)v;
    }
  }
}

__global__ __launch_bounds__(32) void nkf_scan_kernel(ScanP P) {
  // ---- LDS ----
  __shared__ __align__(32) __bf16 wpack[26][32][16];   // packed B-operand weight tiles
  __shared__ float bir[18], bii[18], b1r[18], b1i[18], b2r[4], b2i[4];
  __shared__ float bihr[54], bhhr[54], bihi[54], bhhi[54];
  __shared__ float sAin, sAout;
  __shared__ float hS[4][288];                         // GRU hiddens f32 [16][18]
  __shared__ __align__(16) __bf16 hA[4][16 * 32];      // GRU hiddens bf16 A-staging
  __shared__ __align__(16) __bf16 featAr[16 * 32], featAi[16 * 32];
  __shared__ __align__(16) __bf16 zAr[16 * 32], zAi[16 * 32];
  __shared__ __align__(16) __bf16 yAr[16 * 32], yAi[16 * 32];
  __shared__ __align__(16) __bf16 wAr[16 * 32], wAi[16 * 32];
  __shared__ float hprR[16][4], hprI[16][4];           // h_prior (complex)
  __shared__ float hpoR[16][4], hpoI[16][4];           // h_post  (complex)
  __shared__ float xtR[16][4], xtI[16][4];
  __shared__ float eR[16], eI[16];
  __shared__ float S3[16 * 57], S4[16 * 57];           // gi / gh f32
  __shared__ float kgR[16][4], kgI[16][4];

  const int lane = threadIdx.x;

  // ---- one-time weight prepack into WMMA-B layout ----
  {
    const float* Wp[26]; int Wo[26], Wk[26], Wc0[26];
    int ti = 0;
    auto setT = [&](const float* W, int O, int K, int nt) {
      for (int c = 0; c < nt; ++c) { Wp[ti] = W; Wo[ti] = O; Wk[ti] = K; Wc0[ti] = c * 16; ++ti; }
    };
    setT(P.fcin_rW, 18,  9, 2);  // 0,1
    setT(P.fcin_iW, 18,  9, 2);  // 2,3
    setT(P.grr_wih, 54, 18, 4);  // 4-7
    setT(P.grr_whh, 54, 18, 4);  // 8-11
    setT(P.gri_wih, 54, 18, 4);  // 12-15
    setT(P.gri_whh, 54, 18, 4);  // 16-19
    setT(P.fc1_rW,  18, 18, 2);  // 20,21
    setT(P.fc1_iW,  18, 18, 2);  // 22,23
    setT(P.fc2_rW,   4, 18, 1);  // 24
    setT(P.fc2_iW,   4, 18, 1);  // 25
    int c16 = lane & 15, hf = lane >> 4;
    for (int tt = 0; tt < 26; ++tt) {
      int col = Wc0[tt] + c16;
#pragma unroll
      for (int j = 0; j < 16; ++j) {
        int k = kidx(j, hf);
        float v = (col < Wo[tt] && k < Wk[tt]) ? Wp[tt][col * Wk[tt] + k] : 0.f;
        wpack[tt][lane][j] = (__bf16)v;
      }
    }
    for (int i = lane; i < 18; i += 32) {
      bir[i] = P.fcin_rb[i]; bii[i] = P.fcin_ib[i];
      b1r[i] = P.fc1_rb[i];  b1i[i] = P.fc1_ib[i];
    }
    for (int i = lane; i < 54; i += 32) {
      bihr[i] = P.grr_bih[i]; bhhr[i] = P.grr_bhh[i];
      bihi[i] = P.gri_bih[i]; bhhi[i] = P.gri_bhh[i];
    }
    if (lane < 4) { b2r[lane] = P.fc2_rb[lane]; b2i[lane] = P.fc2_ib[lane]; }
    if (lane == 0) { sAin = P.a_in[0]; sAout = P.a_out[0]; }
    for (int i = lane; i < 4 * 288; i += 32) ((float*)hS)[i] = 0.f;
    for (int i = lane; i < 64; i += 32) {
      ((float*)hprR)[i] = 0.f; ((float*)hprI)[i] = 0.f;
      ((float*)hpoR)[i] = 0.f; ((float*)hpoI)[i] = 0.f;
    }
    // zero bf16 staging (incl. K-padding columns that are never rewritten)
    for (int i = lane; i < 4 * 512; i += 32) ((__bf16*)hA)[i] = (__bf16)0.f;
    for (int i = lane; i < 512; i += 32) {
      featAr[i] = (__bf16)0.f; featAi[i] = (__bf16)0.f;
      zAr[i] = (__bf16)0.f;    zAi[i] = (__bf16)0.f;
      yAr[i] = (__bf16)0.f;    yAi[i] = (__bf16)0.f;
      wAr[i] = (__bf16)0.f;    wAi[i] = (__bf16)0.f;
    }
  }
  __syncthreads();

  const float aIn = sAin, aOut = sAout;
  const int n0 = blockIdx.x * 16;

  for (int t = 0; t < P.T; ++t) {
    if (P.skip[t]) {                                   // wave-uniform branch
      if (lane < 16 && (n0 + lane) < P.N) {
        size_t o = (size_t)(n0 + lane) * P.T + t;
        P.Er[o] = 0.f; P.Ei[o] = 0.f;
      }
      continue;
    }
    // ---- 1. features: xt, e = y - <xt, h_post>, dh = h_post - h_prior ----
    if (lane < 16) {
      int n = n0 + lane;
      bool valid = n < P.N;
      size_t base = (size_t)n * P.T;
      float yr = 0.f, yi = 0.f;
      if (valid) { yr = P.Yr[base + t]; yi = P.Yi[base + t]; }
      float er = yr, ei = yi;
#pragma unroll
      for (int l = 0; l < 4; ++l) {
        int tt = t - 3 + l;
        float xr = 0.f, xi = 0.f;
        if (valid && tt >= 0) { xr = P.Xr[base + tt]; xi = P.Xi[base + tt]; }
        xtR[lane][l] = xr; xtI[lane][l] = xi;
        float hr_ = hpoR[lane][l], hi_ = hpoI[lane][l];
        er -= xr * hr_ - xi * hi_;
        ei -= xr * hi_ + xi * hr_;
        featAr[lane * 32 + l] = (__bf16)xr;
        featAi[lane * 32 + l] = (__bf16)xi;
        featAr[lane * 32 + 5 + l] = (__bf16)(hr_ - hprR[lane][l]);
        featAi[lane * 32 + 5 + l] = (__bf16)(hi_ - hprI[lane][l]);
      }
      featAr[lane * 32 + 4] = (__bf16)er;
      featAi[lane * 32 + 4] = (__bf16)ei;
      eR[lane] = er; eI[lane] = ei;
    }
    __syncthreads();
    // ---- 2. z = PReLU(fc_in(feat)) -> zAr/zAi (bf16 A-staging) ----
    {
      v16bf aR = load_Abf(featAr, lane);
      v16bf aI = load_Abf(featAi, lane);
#pragma unroll
      for (int o = 0; o < 2; ++o) {
        store_bf16(wmma_bf(aR, wpack[0 + o][lane]), zAr, bir, o * 16, 18, lane, aIn);
        store_bf16(wmma_bf(aI, wpack[2 + o][lane]), zAi, bii, o * 16, 18, lane, aIn);
      }
    }
    __syncthreads();
    // ---- 3. GRUs: hrr=R(zr,h0) hir=R(zi,h1) hri=I(zr,h2) hii=I(zi,h3) ----
    {
      const __bf16* gin[4]  = { zAr, zAi, zAr, zAi };
      const int     wih0[4] = { 4, 4, 12, 12 };
      const int     whh0[4] = { 8, 8, 16, 16 };
      const float*  bih[4]  = { bihr, bihr, bihi, bihi };
      const float*  bhh[4]  = { bhhr, bhhr, bhhi, bhhi };
#pragma unroll
      for (int g = 0; g < 4; ++g) {
        v16bf aX = load_Abf(gin[g], lane);
        v16bf aH = load_Abf(hA[g], lane);
#pragma unroll
        for (int o = 0; o < 4; ++o) {
          store_f32(wmma_bf(aX, wpack[wih0[g] + o][lane]), S3, 57, bih[g], o * 16, 54, lane);
          store_f32(wmma_bf(aH, wpack[whh0[g] + o][lane]), S4, 57, bhh[g], o * 16, 54, lane);
        }
        __syncthreads();
        for (int idx = lane; idx < 288; idx += 32) {
          int row = idx / 18, j = idx - row * 18;
          float ir  = S3[row * 57 + j], iz = S3[row * 57 + 18 + j], inn = S3[row * 57 + 36 + j];
          float hrv = S4[row * 57 + j], hz = S4[row * 57 + 18 + j], hn  = S4[row * 57 + 36 + j];
          float rg = 1.f / (1.f + __expf(-(ir + hrv)));
          float zg = 1.f / (1.f + __expf(-(iz + hz)));
          float ng = tanhf(inn + rg * hn);
          float hnew = (1.f - zg) * ng + zg * hS[g][idx];
          hS[g][idx] = hnew;
          hA[g][row * 32 + j] = (__bf16)hnew;
        }
        __syncthreads();
      }
    }
    // ---- 4. y = (hrr - hii, hri + hir) -> yAr/yAi (bf16 A-staging) ----
    for (int idx = lane; idx < 288; idx += 32) {
      int row = idx / 18, j = idx - row * 18;
      yAr[row * 32 + j] = (__bf16)(hS[0][idx] - hS[3][idx]);
      yAi[row * 32 + j] = (__bf16)(hS[2][idx] + hS[1][idx]);
    }
    __syncthreads();
    // ---- 5. w = PReLU(fc1(y)) -> wAr/wAi ----
    {
      v16bf aR = load_Abf(yAr, lane);
      v16bf aI = load_Abf(yAi, lane);
#pragma unroll
      for (int o = 0; o < 2; ++o) {
        store_bf16(wmma_bf(aR, wpack[20 + o][lane]), wAr, b1r, o * 16, 18, lane, aOut);
        store_bf16(wmma_bf(aI, wpack[22 + o][lane]), wAi, b1i, o * 16, 18, lane, aOut);
      }
    }
    __syncthreads();
    // ---- 6. kg = fc2(w) -> kgR/kgI f32 [16][4] ----
    {
      v16bf aR = load_Abf(wAr, lane);
      v16bf aI = load_Abf(wAi, lane);
      store_f32(wmma_bf(aR, wpack[24][lane]), &kgR[0][0], 4, b2r, 0, 4, lane);
      store_f32(wmma_bf(aI, wpack[25][lane]), &kgI[0][0], 4, b2i, 0, 4, lane);
    }
    __syncthreads();
    // ---- 7. Kalman update + echo ----
    if (lane < 16) {
      int n = n0 + lane;
      float er = eR[lane], ei = eI[lane];
      float accR = 0.f, accI = 0.f;
#pragma unroll
      for (int l = 0; l < 4; ++l) {
        float kr = kgR[lane][l], ki = kgI[lane][l];
        float hr_ = hpoR[lane][l], hi_ = hpoI[lane][l];     // hp = old h_post
        float hqr = hr_ + kr * er - ki * ei;
        float hqi = hi_ + kr * ei + ki * er;
        hprR[lane][l] = hr_; hprI[lane][l] = hi_;           // h_prior <- hp
        hpoR[lane][l] = hqr; hpoI[lane][l] = hqi;           // h_post  <- hq
        float xr = xtR[lane][l], xi = xtI[lane][l];
        accR += xr * hqr - xi * hqi;
        accI += xr * hqi + xi * hqr;
      }
      if (n < P.N) {
        size_t o = (size_t)n * P.T + t;
        P.Er[o] = accR; P.Ei[o] = accI;
      }
    }
    __syncthreads();
  }
}

// ---------------------------------------------------------------- ISTFT frames
__global__ __launch_bounds__(256) void iframes_kernel(
    const float* __restrict__ Yr, const float* __restrict__ Yi,
    const float* __restrict__ Er, const float* __restrict__ Ei,
    float* __restrict__ frames, int T) {
  int t = blockIdx.x, b = blockIdx.y, tid = threadIdx.x;
  __shared__ float re[1024], im[1024];
  for (int f = tid; f <= 512; f += 256) {
    size_t n = (size_t)(b * NBINS + f) * T + t;
    re[f] = Yr[n] - Er[n];
    im[f] = Yi[n] - Ei[n];
  }
  __syncthreads();
  for (int i = tid; i < 1024; i += 256) {
    if (i > 512) { re[i] = re[1024 - i]; im[i] = -im[1024 - i]; }
  }
  __syncthreads();
  fft1024(re, im, true, tid);
  for (int i = tid; i < 1024; i += 256)
    frames[((size_t)(b * T + t)) * 1024 + i] = re[i] * (1.0f / 1024.0f) * hannw(i);
}

// ---------------------------------------------------------------- overlap-add
__global__ void ola_kernel(const float* __restrict__ frames, float* __restrict__ out,
                           int T, int LEN, int B) {
  int idx = blockIdx.x * blockDim.x + threadIdx.x;
  if (idx >= B * LEN) return;
  int b = idx / LEN, s = idx - b * LEN;
  int j = s + 512;
  int tlo = (j >= 1024) ? ((j - 1023 + 255) >> 8) : 0;
  int thi = j >> 8; if (thi > T - 1) thi = T - 1;
  float acc = 0.f, env = 0.f;
  for (int t = tlo; t <= thi; ++t) {
    int k = j - t * HOP;
    float wv = hannw(k);
    acc += frames[((size_t)(b * T + t)) * 1024 + k];
    env += wv * wv;
  }
  out[idx] = acc / (env > 1e-11f ? env : 1.0f);
}

// ---------------------------------------------------------------- launch
extern "C" void kernel_launch(void* const* d_in, const int* in_sizes, int n_in,
                              void* d_out, int out_size, void* d_ws, size_t ws_size,
                              hipStream_t stream) {
  const int B   = 8;
  const int LEN = in_sizes[0] / B;          // 160000
  const int T   = 1 + LEN / HOP;            // 626
  const int N   = B * NBINS;                // 4104
  const int tiles = (N + 15) / 16;          // 257
  const size_t NP = (size_t)tiles * 16;

  const float* x = (const float*)d_in[0];
  const float* y = (const float*)d_in[1];

  char* p = (char*)d_ws;
  auto carve = [&](size_t bytes) -> char* {
    char* r = p; p += (bytes + 255) & ~(size_t)255; return r;
  };
  float* Xr = (float*)carve(NP * T * sizeof(float));
  float* Xi = (float*)carve(NP * T * sizeof(float));
  float* Yr = (float*)carve(NP * T * sizeof(float));
  float* Yi = (float*)carve(NP * T * sizeof(float));
  float* Er = (float*)carve(NP * T * sizeof(float));
  float* Ei = (float*)carve(NP * T * sizeof(float));
  int*   sk = (int*)carve((size_t)T * sizeof(int));
  float* fr = (float*)carve((size_t)B * T * NFFT * sizeof(float));

  stft_kernel<<<dim3(T, B, 2), 256, 0, stream>>>(x, y, Xr, Xi, Yr, Yi, T, LEN);
  skip_kernel<<<T, 256, 0, stream>>>(Xr, Xi, sk, T, N);

  // params flattened in jax pytree (alphabetical) order after x, y
  ScanP P;
  P.a_in   = (const float*)d_in[2];  P.a_out  = (const float*)d_in[3];
  P.fc1_iW = (const float*)d_in[4];  P.fc1_ib = (const float*)d_in[5];
  P.fc1_rW = (const float*)d_in[6];  P.fc1_rb = (const float*)d_in[7];
  P.fc2_iW = (const float*)d_in[8];  P.fc2_ib = (const float*)d_in[9];
  P.fc2_rW = (const float*)d_in[10]; P.fc2_rb = (const float*)d_in[11];
  P.fcin_iW = (const float*)d_in[12]; P.fcin_ib = (const float*)d_in[13];
  P.fcin_rW = (const float*)d_in[14]; P.fcin_rb = (const float*)d_in[15];
  P.gri_bhh = (const float*)d_in[16]; P.gri_bih = (const float*)d_in[17];
  P.gri_whh = (const float*)d_in[18]; P.gri_wih = (const float*)d_in[19];
  P.grr_bhh = (const float*)d_in[20]; P.grr_bih = (const float*)d_in[21];
  P.grr_whh = (const float*)d_in[22]; P.grr_wih = (const float*)d_in[23];
  P.Xr = Xr; P.Xi = Xi; P.Yr = Yr; P.Yi = Yi;
  P.Er = Er; P.Ei = Ei; P.skip = sk; P.T = T; P.N = N;
  nkf_scan_kernel<<<tiles, 32, 0, stream>>>(P);

  iframes_kernel<<<dim3(T, B), 256, 0, stream>>>(Yr, Yi, Er, Ei, fr, T);
  ola_kernel<<<(B * LEN + 255) / 256, 256, 0, stream>>>(fr, (float*)d_out, T, LEN, B);
}
